// TestModel_45827301048693
// MI455X (gfx1250) — 8-bit matrix operand order repaired and checked by reading, compiled, not run on hardware
//
#include <hip/hip_runtime.h>
#include <stdint.h>

#define TOKENS   8192
#define HIDDEN   4096
#define KDIM     HIDDEN
#define FP8_MAX  448.0f
#define EPS      1e-6f

typedef __attribute__((ext_vector_type(16))) int          v16i;
typedef __attribute__((ext_vector_type(8)))  float        v8f;
typedef __attribute__((ext_vector_type(4)))  int          v4i;
typedef __attribute__((ext_vector_type(4)))  float        v4f;
typedef __attribute__((ext_vector_type(4)))  unsigned int v4u;

#define AS1 __attribute__((address_space(1)))
#define AS3 __attribute__((address_space(3)))

#if defined(__has_builtin)
#  if __has_builtin(__builtin_amdgcn_global_load_async_to_lds_b128)
#    define HAVE_ASYNC_LDS 1
#  endif
#  if __has_builtin(__builtin_amdgcn_s_wait_asynccnt)
#    define HAVE_WAIT_ASYNC 1
#  endif
#endif
#ifndef HAVE_ASYNC_LDS
#  define HAVE_ASYNC_LDS 0
#endif
#ifndef HAVE_WAIT_ASYNC
#  define HAVE_WAIT_ASYNC 0
#endif

#if HAVE_ASYNC_LDS
#  if HAVE_WAIT_ASYNC
#    define WAIT_ASYNC0() __builtin_amdgcn_s_wait_asynccnt(0)
#  else
#    define WAIT_ASYNC0() asm volatile("s_wait_asynccnt 0" ::: "memory")
#  endif
#endif

__device__ __forceinline__ unsigned f32_to_e4m3(float x) {
  float c = fminf(fmaxf(x, -FP8_MAX), FP8_MAX);
  unsigned u = __float_as_uint(c);
  unsigned s = (u >> 31) << 7;
  int e = (int)((u >> 23) & 0xFFu);
  unsigned m = u & 0x7FFFFFu;
  int ee = e - 127 + 7;
  if (ee <= 0) {
    float a = fabsf(c) * 512.0f;
    int q = (int)(a + 0.5f);
    if (q > 7) q = 7;
    return s | (unsigned)q;
  }
  unsigned mant = m >> 20;
  unsigned rem  = m & 0xFFFFFu;
  if (rem > 0x80000u || (rem == 0x80000u && (mant & 1u))) {
    mant++;
    if (mant == 8u) { mant = 0u; ee++; }
  }
  if (ee >= 16) return s | 0x7Eu;
  return s | ((unsigned)ee << 3) | mant;
}

__device__ __forceinline__ unsigned pack4_e4m3(v4f v, float scale) {
  return  f32_to_e4m3(v[0] * scale)
       | (f32_to_e4m3(v[1] * scale) << 8)
       | (f32_to_e4m3(v[2] * scale) << 16)
       | (f32_to_e4m3(v[3] * scale) << 24);
}

__global__ __launch_bounds__(256)
void norm_quant_kernel(const float* __restrict__ in,
                       const float* __restrict__ gamma,
                       float* __restrict__ resid_out,
                       unsigned char* __restrict__ yq,
                       float* __restrict__ a_scale,
                       int do_relu, int write_resid) {
  __shared__ float sred0[8];
  __shared__ float sred1[8];
  const int row  = blockIdx.x;
  const int tid  = threadIdx.x;
  const int lane = tid & 31;
  const int wave = tid >> 5;
  const float* xr = in + (size_t)row * HIDDEN;

  v4f vals[4];
  float ss = 0.f;
#pragma unroll
  for (int p = 0; p < 4; ++p) {
    const int vidx = tid + p * 256;
    v4f v = *(const v4f*)(xr + (size_t)vidx * 4);
    if (do_relu) {
      v[0] = fmaxf(v[0], 0.f); v[1] = fmaxf(v[1], 0.f);
      v[2] = fmaxf(v[2], 0.f); v[3] = fmaxf(v[3], 0.f);
    }
    vals[p] = v;
    ss += v[0]*v[0] + v[1]*v[1] + v[2]*v[2] + v[3]*v[3];
    if (write_resid)
      *(v4f*)(resid_out + (size_t)row * HIDDEN + (size_t)vidx * 4) = v;
  }
  for (int o = 16; o > 0; o >>= 1) ss += __shfl_xor(ss, o, 32);
  if (lane == 0) sred0[wave] = ss;
  __syncthreads();
  if (tid < 32) {
    float t = (lane < 8) ? sred0[lane] : 0.f;
    for (int o = 4; o > 0; o >>= 1) t += __shfl_xor(t, o, 32);
    if (lane == 0) sred0[0] = t;
  }
  __syncthreads();
  const float inv_rms = rsqrtf(sred0[0] / (float)HIDDEN + EPS);

  float amax = 0.f;
#pragma unroll
  for (int p = 0; p < 4; ++p) {
    const int vidx = tid + p * 256;
    v4f g = *(const v4f*)(gamma + (size_t)vidx * 4);
    v4f y = vals[p];
    y[0] *= inv_rms * g[0]; y[1] *= inv_rms * g[1];
    y[2] *= inv_rms * g[2]; y[3] *= inv_rms * g[3];
    vals[p] = y;
    amax = fmaxf(amax, fmaxf(fmaxf(fabsf(y[0]), fabsf(y[1])),
                             fmaxf(fabsf(y[2]), fabsf(y[3]))));
  }
  for (int o = 16; o > 0; o >>= 1) amax = fmaxf(amax, __shfl_xor(amax, o, 32));
  if (lane == 0) sred1[wave] = amax;
  __syncthreads();
  if (tid < 32) {
    float t = (lane < 8) ? sred1[lane] : 0.f;
    for (int o = 4; o > 0; o >>= 1) t = fmaxf(t, __shfl_xor(t, o, 32));
    if (lane == 0) sred1[0] = t;
  }
  __syncthreads();
  const float s = fmaxf(sred1[0] / FP8_MAX, 1e-12f);
  const float inv_s = 1.f / s;
  if (tid == 0) a_scale[row] = s;
#pragma unroll
  for (int p = 0; p < 4; ++p) {
    const int vidx = tid + p * 256;
    *(unsigned*)(yq + (size_t)row * HIDDEN + (size_t)vidx * 4) =
        pack4_e4m3(vals[p], inv_s);
  }
}

__global__ __launch_bounds__(256)
void norm_out_kernel(const float* __restrict__ in,
                     const float* __restrict__ gamma,
                     float* __restrict__ out) {
  __shared__ float sred0[8];
  const int row  = blockIdx.x;
  const int tid  = threadIdx.x;
  const int lane = tid & 31;
  const int wave = tid >> 5;
  const float* xr = in + (size_t)row * HIDDEN;

  v4f vals[4];
  float ss = 0.f;
#pragma unroll
  for (int p = 0; p < 4; ++p) {
    const int vidx = tid + p * 256;
    v4f v = *(const v4f*)(xr + (size_t)vidx * 4);
    vals[p] = v;
    ss += v[0]*v[0] + v[1]*v[1] + v[2]*v[2] + v[3]*v[3];
  }
  for (int o = 16; o > 0; o >>= 1) ss += __shfl_xor(ss, o, 32);
  if (lane == 0) sred0[wave] = ss;
  __syncthreads();
  if (tid < 32) {
    float t = (lane < 8) ? sred0[lane] : 0.f;
    for (int o = 4; o > 0; o >>= 1) t += __shfl_xor(t, o, 32);
    if (lane == 0) sred0[0] = t;
  }
  __syncthreads();
  const float inv_rms = rsqrtf(sred0[0] / (float)HIDDEN + EPS);
#pragma unroll
  for (int p = 0; p < 4; ++p) {
    const int vidx = tid + p * 256;
    v4f g = *(const v4f*)(gamma + (size_t)vidx * 4);
    v4f y = vals[p];
    y[0] *= inv_rms * g[0]; y[1] *= inv_rms * g[1];
    y[2] *= inv_rms * g[2]; y[3] *= inv_rms * g[3];
    *(v4f*)(out + (size_t)row * HIDDEN + (size_t)vidx * 4) = y;
  }
}

__global__ __launch_bounds__(256)
void quant_w_kernel(const float* __restrict__ W,
                    const float* __restrict__ wscale,
                    unsigned char* __restrict__ wq) {
  const size_t gid = (size_t)blockIdx.x * blockDim.x + threadIdx.x;
  const float inv = 1.0f / wscale[0];
  const v4f* src = (const v4f*)W + gid * 4;
  v4u o;
#pragma unroll
  for (int c = 0; c < 4; ++c) o[c] = pack4_e4m3(src[c], inv);
  ((v4u*)wq)[gid] = o;
}

#define LDS_COL_STRIDE 144
#define LDS_BUF_BYTES  (64 * LDS_COL_STRIDE)

__global__ __launch_bounds__(256)
void gemm_fp8_resid_kernel(const unsigned char* __restrict__ yq,
                           const unsigned char* __restrict__ wq,
                           const float* __restrict__ a_scale,
                           const float* __restrict__ w_scale,
                           float* __restrict__ resid) {
  __shared__ unsigned char ldsB[2 * LDS_BUF_BYTES];

  const int tid    = threadIdx.x;
  const int lane   = tid & 31;
  const int wave   = tid >> 5;
  const int lane16 = lane & 15;
  const int half   = lane >> 4;
  const int n0     = blockIdx.x * 64;
  const int trow   = blockIdx.y * 256 + wave * 32;

  v8f acc[2][4] = {};

  const int colS  = tid >> 2;
  const int partS = tid & 3;
  const int lofs  = colS * LDS_COL_STRIDE + partS * 32;
  const unsigned char* gW = wq + (size_t)(n0 + colS) * KDIM + partS * 32;

  const unsigned char* gA0 = yq + (size_t)(trow + lane16) * KDIM + half * 64;
  const unsigned char* gA1 = gA0 + (size_t)16 * KDIM;

#if HAVE_ASYNC_LDS
#  define STAGE_B(bufofs, kk)                                                   \
    do {                                                                        \
      __builtin_amdgcn_global_load_async_to_lds_b128(                           \
          (AS1 v4i*)(gW + (kk)),      (AS3 v4i*)(ldsB + (bufofs) + lofs), 0, 0);\
      __builtin_amdgcn_global_load_async_to_lds_b128(                           \
          (AS1 v4i*)(gW + (kk) + 16), (AS3 v4i*)(ldsB + (bufofs) + lofs + 16),  \
          0, 0);                                                                \
    } while (0)
#else
#  define STAGE_B(bufofs, kk)                                                   \
    do {                                                                        \
      const v4u* s_ = (const v4u*)(gW + (kk));                                  \
      v4u*       d_ = (v4u*)(ldsB + (bufofs) + lofs);                           \
      d_[0] = s_[0];                                                            \
      d_[1] = s_[1];                                                            \
    } while (0)
#  define WAIT_ASYNC0() ((void)0)
#endif

  STAGE_B(0, 0);
  WAIT_ASYNC0();
  __syncthreads();

  for (int k = 0; k < KDIM; k += 128) {
    const int curOfs = ((k >> 7) & 1) ? LDS_BUF_BYTES : 0;
    const int nxtOfs = curOfs ^ LDS_BUF_BYTES;

    if (k + 128 < KDIM) {
      STAGE_B(nxtOfs, k + 128);
      __builtin_prefetch(gW + k + 256 < gW + KDIM ? gW + k + 256 : gW + k, 0, 0);
    }

    v16i a0 = *(const v16i*)(gA0 + k);
    v16i a1 = *(const v16i*)(gA1 + k);

#pragma unroll
    for (int j = 0; j < 4; ++j) {
      const unsigned char* cb =
          ldsB + curOfs + (j * 16 + lane16) * LDS_COL_STRIDE + half * 64;
      v16i b;
#pragma unroll
      for (int c = 0; c < 4; ++c) {
        v4i t = *(const v4i*)(cb + c * 16);
        b[4*c + 0] = t[0]; b[4*c + 1] = t[1];
        b[4*c + 2] = t[2]; b[4*c + 3] = t[3];
      }
      acc[0][j] = __builtin_amdgcn_wmma_f32_16x16x128_fp8_fp8(
          a0, b, (short)0, acc[0][j], false, false);
      acc[1][j] = __builtin_amdgcn_wmma_f32_16x16x128_fp8_fp8(
          a1, b, (short)0, acc[1][j], false, false);
    }

    WAIT_ASYNC0();
    __syncthreads();
  }

  const float sw = w_scale[0];
#pragma unroll
  for (int m = 0; m < 2; ++m) {
#pragma unroll
    for (int j = 0; j < 4; ++j) {
      const int n = n0 + j * 16 + lane16;
#pragma unroll
      for (int r = 0; r < 8; ++r) {
        const int t = trow + m * 16 + half * 8 + r;
        const size_t idx = (size_t)t * HIDDEN + n;
        resid[idx] += a_scale[t] * sw * acc[m][j][r];
      }
    }
  }
}

extern "C" void kernel_launch(void* const* d_in, const int* in_sizes, int n_in,
                              void* d_out, int out_size, void* d_ws, size_t ws_size,
                              hipStream_t stream) {
  const float* x       = (const float*)d_in[0];
  const float* norm_w  = (const float*)d_in[1];
  const float* Ws      = (const float*)d_in[2];
  const float* wscales = (const float*)d_in[3];
  float* out = (float*)d_out;

  char* ws = (char*)d_ws;
  float* resid       = (float*)ws;         ws += (size_t)TOKENS * HIDDEN * sizeof(float);
  unsigned char* yq  = (unsigned char*)ws; ws += (size_t)TOKENS * HIDDEN;
  unsigned char* wq  = (unsigned char*)ws; ws += (size_t)HIDDEN * HIDDEN;
  float* a_scale     = (float*)ws;

  dim3 blk(256);
  norm_quant_kernel<<<TOKENS, blk, 0, stream>>>(x, norm_w, resid, yq, a_scale, 1, 1);

  for (int i = 0; i < 3; ++i) {
    quant_w_kernel<<<(HIDDEN * (size_t)HIDDEN) / (256 * 16), blk, 0, stream>>>(
        Ws + (size_t)i * HIDDEN * HIDDEN, wscales + i, wq);

    dim3 grid(HIDDEN / 64, TOKENS / 256);
    gemm_fp8_resid_kernel<<<grid, blk, 0, stream>>>(yq, wq, a_scale, wscales + i, resid);

    if (i < 2)
      norm_quant_kernel<<<TOKENS, blk, 0, stream>>>(
          resid, norm_w + (size_t)(i + 1) * HIDDEN, nullptr, yq, a_scale, 0, 0);
  }
  norm_out_kernel<<<TOKENS, blk, 0, stream>>>(resid, norm_w + (size_t)3 * HIDDEN, out);
}
